// Model_25048249270414
// MI455X (gfx1250) — matrix operand order repaired and checked by reading, not compiled, not run on hardware
//
#include <hip/hip_runtime.h>
#include <hip/hip_bf16.h>

// ---------------- problem constants ----------------
#define BB   4
#define NN   512
#define MM   32
#define CC   512
#define TT   8
#define HF   14
#define WF   14
#define KK   80
#define PP   7
#define HID  1024
#define KF   25088        // C*P*P
#define ROWS 2048         // B*N

typedef __attribute__((ext_vector_type(16))) __bf16 v16bf;
typedef __attribute__((ext_vector_type(8)))  float  v8f;

union Frag16 {            // 8 VGPRs = 16 bf16 halves
  v16bf v;
  unsigned short u[16];
  unsigned int   w[8];
};

__device__ __forceinline__ unsigned short f2bf(float f) {
  unsigned int u = __float_as_uint(f);
  u += 0x7fffu + ((u >> 16) & 1u);          // round-to-nearest-even
  return (unsigned short)(u >> 16);
}

// ---------------- 1) temporal mean: (B,C,T,H,W) -> BHWC (B,H,W,C) ----------------
__global__ __launch_bounds__(256) void k_tmean(const float* __restrict__ feature,
                                               float* __restrict__ feat) {
  int o = blockIdx.x * 256 + threadIdx.x;
  if (o >= BB * HF * WF * CC) return;
  int c = o & (CC - 1);
  int s = o >> 9;
  int x = s % WF; s /= WF;
  int y = s % HF;
  int b = s / HF;
  const float* p = feature + ((size_t)(b * CC + c) * TT) * (HF * WF) + y * WF + x;
  float acc = 0.f;
#pragma unroll
  for (int t = 0; t < TT; ++t) acc += p[t * HF * WF];
  feat[o] = acc * (1.0f / TT);
}

// ---------------- 2a) W1 f32 (25088x1024) -> bf16 TRANSPOSED (1024x25088) ----------------
// 32x32 LDS tile transpose: coalesced f32 reads, coalesced bf16 writes.
__global__ __launch_bounds__(256) void k_cvtT1(const float* __restrict__ in,
                                               unsigned short* __restrict__ out) {
  __shared__ unsigned short tile[32][33];
  const int tx = threadIdx.x;          // 0..31
  const int ty = threadIdx.y;          // 0..7
  const int r0 = blockIdx.y * 32;      // K block (rows of W1)
  const int c0 = blockIdx.x * 32;      // N block (cols of W1)
#pragma unroll
  for (int i = ty; i < 32; i += 8)
    tile[i][tx] = f2bf(in[(size_t)(r0 + i) * HID + (c0 + tx)]);
  __syncthreads();
#pragma unroll
  for (int i = ty; i < 32; i += 8)
    out[(size_t)(c0 + i) * KF + (r0 + tx)] = tile[tx][i];
}

// ---------------- 2b) W2 f32 (1024x80) -> bf16 TRANSPOSED (80x1024) ----------------
__global__ __launch_bounds__(256) void k_cvtT2(const float* __restrict__ in,
                                               unsigned short* __restrict__ out) {
  int id = blockIdx.x * 256 + threadIdx.x;     // 0..81919
  if (id >= HID * KK) return;
  int k = id & (HID - 1);                      // fastest -> coalesced writes
  int n = id >> 10;
  out[(size_t)n * HID + k] = f2bf(in[(size_t)k * KK + n]);
}

// ---------------- 3) IoU max / argmax per ROI ----------------
__global__ __launch_bounds__(256) void k_iou(const float* __restrict__ prop,
                                             const float* __restrict__ gt,
                                             float* __restrict__ maxiou,
                                             int* __restrict__ assign) {
  int id = blockIdx.x * 256 + threadIdx.x;
  if (id >= ROWS) return;
  int b = id >> 9;
  float ax1 = prop[id * 4 + 0], ay1 = prop[id * 4 + 1];
  float ax2 = prop[id * 4 + 2], ay2 = prop[id * 4 + 3];
  float aa = (ax2 - ax1) * (ay2 - ay1);
  float best = -1.f; int bi = 0;
#pragma unroll 4
  for (int m = 0; m < MM; ++m) {
    const float* g = gt + ((size_t)b * MM + m) * 4;
    float lx = fmaxf(ax1, g[0]), ly = fmaxf(ay1, g[1]);
    float rx = fminf(ax2, g[2]), ry = fminf(ay2, g[3]);
    float w = fmaxf(rx - lx, 0.f), h = fmaxf(ry - ly, 0.f);
    float inter = w * h;
    float ab = (g[2] - g[0]) * (g[3] - g[1]);
    float iou = inter / (aa + ab - inter + 1e-8f);
    if (iou > best) { best = iou; bi = m; }   // first-max like jnp.argmax
  }
  maxiou[id] = best;
  assign[id] = bi;
}

// ---------------- 4) global any-fg flag ----------------
__global__ __launch_bounds__(256) void k_flag(const float* __restrict__ maxiou,
                                              int* __restrict__ flag) {
  __shared__ int s;
  if (threadIdx.x == 0) s = 0;
  __syncthreads();
  int any = 0;
  for (int i = threadIdx.x; i < ROWS; i += 256) any |= (maxiou[i] >= 0.85f) ? 1 : 0;
  if (any) atomicOr(&s, 1);
  __syncthreads();
  if (threadIdx.x == 0) *flag = s;
}

// ---------------- 5) ROI-align bilinear pool -> bf16 A-matrix ----------------
__global__ __launch_bounds__(256) void k_roipool(const float* __restrict__ feat,
                                                 const float* __restrict__ boxes,
                                                 unsigned short* __restrict__ pool) {
  int row = blockIdx.x;                 // 0..2047
  int b = row >> 9;
  __shared__ int   sy0[PP], sy1[PP], sx0[PP], sx1[PP];
  __shared__ float swy[PP], swx[PP];
  if (threadIdx.x < PP) {
    int p = threadIdx.x;
    const float sc = (float)HF / 224.0f;          // 0.0625
    float bx1 = boxes[row * 4 + 0] * sc, by1 = boxes[row * 4 + 1] * sc;
    float bx2 = boxes[row * 4 + 2] * sc, by2 = boxes[row * 4 + 3] * sc;
    float g = ((float)p + 0.5f) / (float)PP;
    float y = by1 + g * (by2 - by1) - 0.5f;
    float x = bx1 + g * (bx2 - bx1) - 0.5f;
    float yf = floorf(y), xf = floorf(x);
    swy[p] = fminf(fmaxf(y - yf, 0.f), 1.f);
    swx[p] = fminf(fmaxf(x - xf, 0.f), 1.f);
    sy0[p] = min(max((int)yf,     0), HF - 1);
    sy1[p] = min(max((int)yf + 1, 0), HF - 1);
    sx0[p] = min(max((int)xf,     0), WF - 1);
    sx1[p] = min(max((int)xf + 1, 0), WF - 1);
  }
  __syncthreads();
  const float* fb = feat + (size_t)b * HF * WF * CC;   // BHWC
  unsigned short* out = pool + (size_t)row * KF;
  for (int k = threadIdx.x; k < KF; k += 256) {
    int c = k / 49;
    int cell = k - c * 49;
    int py = cell / PP, px = cell - py * PP;
    float wy = swy[py], wx = swx[px];
    int y0 = sy0[py], y1 = sy1[py], x0 = sx0[px], x1 = sx1[px];
    float v00 = fb[(y0 * WF + x0) * CC + c];
    float v01 = fb[(y0 * WF + x1) * CC + c];
    float v10 = fb[(y1 * WF + x0) * CC + c];
    float v11 = fb[(y1 * WF + x1) * CC + c];
    float v = v00 * (1.f - wy) * (1.f - wx) + v01 * (1.f - wy) * wx +
              v10 * wy * (1.f - wx)        + v11 * wy * wx;
    out[k] = f2bf(v);
  }
}

// ---------------- 6) FC1: (2048 x 25088) x (25088 x 1024), bf16 WMMA, ReLU ----------------
// Block tile 128x128, 8 wave32s in 2x4; wave tile 64x32 = 8 x wmma per K-step.
// B tile stored K-contiguous per column (from pre-transposed W1^T) so every
// fragment is two ds_load_b128 — no u16 gathers, no v_mov_b16 packing.
#define AST 40   // LDS A row stride (ushorts): 80B, 16B-aligned
#define BST 40   // LDS B^T col stride (ushorts): 80B, 16B-aligned
__global__ __launch_bounds__(256) void k_gemm1(const unsigned short* __restrict__ Ap,
                                               const unsigned short* __restrict__ BwT,
                                               const float* __restrict__ bias,
                                               unsigned short* __restrict__ Hout) {
  __shared__ unsigned short As[128 * AST];   // 10 KB
  __shared__ unsigned short Bt[128 * BST];   // 10 KB
  const int tid  = threadIdx.x;
  const int lane = tid & 31;
  const int wv   = tid >> 5;
  const int wm   = wv & 1;          // 2 m-subtiles of 64
  const int wn   = wv >> 1;         // 4 n-subtiles of 32
  const int rowBase = blockIdx.y * 128;
  const int colBase = blockIdx.x * 128;

  v8f acc[4][2] = {};

  // staging: thread t covers (row/col = t>>1, K-half = (t&1)*16), 32B each
  const int sr = tid >> 1;
  const int sk = (tid & 1) * 16;

  const int l15   = lane & 15;
  const int ahalf = (lane < 16) ? 0 : 8;     // A K-half select
  const int bhalf = (lane < 16) ? 0 : 8;     // B K-half select

  const unsigned short* aSrc = Ap  + (size_t)(rowBase + sr) * KF + sk;
  const unsigned short* bSrc = BwT + (size_t)(colBase + sr) * KF + sk;

  uint4 ra0, ra1, rb0, rb1;
  ra0 = *(const uint4*)(aSrc);     ra1 = *(const uint4*)(aSrc + 8);
  rb0 = *(const uint4*)(bSrc);     rb1 = *(const uint4*)(bSrc + 8);

  for (int kb = 0; kb < KF; kb += 32) {
    // commit staged regs to LDS
    *(uint4*)(&As[sr * AST + sk])     = ra0;
    *(uint4*)(&As[sr * AST + sk + 8]) = ra1;
    *(uint4*)(&Bt[sr * BST + sk])     = rb0;
    *(uint4*)(&Bt[sr * BST + sk + 8]) = rb1;
    __syncthreads();

    // prefetch next K-slab while WMMAs run
    if (kb + 32 < KF) {
      ra0 = *(const uint4*)(aSrc + kb + 32);
      ra1 = *(const uint4*)(aSrc + kb + 40);
      rb0 = *(const uint4*)(bSrc + kb + 32);
      rb1 = *(const uint4*)(bSrc + kb + 40);
    }

    Frag16 af[4], bf[2];
#pragma unroll
    for (int mi = 0; mi < 4; ++mi) {
      const int m = wm * 64 + mi * 16 + l15;
      const unsigned int* p0 = (const unsigned int*)&As[m * AST + ahalf];
      const unsigned int* p1 = (const unsigned int*)&As[m * AST + 16 + ahalf];
#pragma unroll
      for (int j = 0; j < 4; ++j) { af[mi].w[j] = p0[j]; af[mi].w[4 + j] = p1[j]; }
    }
#pragma unroll
    for (int ni = 0; ni < 2; ++ni) {
      const int n = wn * 32 + ni * 16 + l15;
      const unsigned int* p0 = (const unsigned int*)&Bt[n * BST + bhalf];
      const unsigned int* p1 = (const unsigned int*)&Bt[n * BST + 16 + bhalf];
#pragma unroll
      for (int j = 0; j < 4; ++j) { bf[ni].w[j] = p0[j]; bf[ni].w[4 + j] = p1[j]; }
    }
#pragma unroll
    for (int mi = 0; mi < 4; ++mi)
#pragma unroll
      for (int ni = 0; ni < 2; ++ni)
        acc[mi][ni] = __builtin_amdgcn_wmma_f32_16x16x32_bf16(
            false, af[mi].v, false, bf[ni].v, (short)0, acc[mi][ni], false, false);
    __syncthreads();
  }

  // epilogue: bias + ReLU -> bf16
#pragma unroll
  for (int mi = 0; mi < 4; ++mi)
#pragma unroll
    for (int ni = 0; ni < 2; ++ni) {
      int col = colBase + wn * 32 + ni * 16 + l15;
      float bv = bias[col];
#pragma unroll
      for (int i = 0; i < 8; ++i) {
        int row = rowBase + wm * 64 + mi * 16 + i + ((lane < 16) ? 0 : 8);
        float v = acc[mi][ni][i] + bv;
        v = v > 0.f ? v : 0.f;
        Hout[(size_t)row * HID + col] = f2bf(v);
      }
    }
}

// ---------------- 7) FC2: (2048 x 1024) x (1024 x 80), bf16 WMMA ----------------
// 640 wave-tiles of 16x16 (128 m-tiles x 5 n-tiles); W2^T is L2-resident ->
// direct global fragments, all 16B contiguous loads.
__global__ __launch_bounds__(256) void k_gemm2(const unsigned short* __restrict__ H,
                                               const unsigned short* __restrict__ W2T,
                                               const float* __restrict__ b2,
                                               float* __restrict__ logits) {
  const int gw   = blockIdx.x * 8 + (threadIdx.x >> 5);  // 0..639 exactly
  const int lane = threadIdx.x & 31;
  const int mt = gw / 5, nt = gw % 5;
  const int l15 = lane & 15;
  const int m = mt * 16 + l15;
  const int n = nt * 16 + l15;
  const int ahalf = (lane < 16) ? 0 : 8;
  const int bhalf = (lane < 16) ? 0 : 8;

  const unsigned short* hRow = H   + (size_t)m * HID;
  const unsigned short* wRow = W2T + (size_t)n * HID;

  v8f acc = {};
  for (int kb = 0; kb < HID; kb += 32) {
    Frag16 a, b;
    {
      const unsigned int* p0 = (const unsigned int*)(hRow + kb + ahalf);
      const unsigned int* p1 = (const unsigned int*)(hRow + kb + 16 + ahalf);
#pragma unroll
      for (int j = 0; j < 4; ++j) { a.w[j] = p0[j]; a.w[4 + j] = p1[j]; }
    }
    {
      const unsigned int* p0 = (const unsigned int*)(wRow + kb + bhalf);
      const unsigned int* p1 = (const unsigned int*)(wRow + kb + 16 + bhalf);
#pragma unroll
      for (int j = 0; j < 4; ++j) { b.w[j] = p0[j]; b.w[4 + j] = p1[j]; }
    }
    acc = __builtin_amdgcn_wmma_f32_16x16x32_bf16(
        false, a.v, false, b.v, (short)0, acc, false, false);
  }
  float bv = b2[n];
#pragma unroll
  for (int i = 0; i < 8; ++i) {
    int row = mt * 16 + i + ((lane < 16) ? 0 : 8);
    logits[(size_t)row * KK + n] = acc[i] + bv;
  }
}

// ---------------- 8) BCE loss + fg-masked per-batch mean ----------------
__global__ __launch_bounds__(256) void k_loss(const float* __restrict__ logits,
                                              const float* __restrict__ gtc,
                                              const int* __restrict__ assign,
                                              const float* __restrict__ maxiou,
                                              const int* __restrict__ flag,
                                              float* __restrict__ out) {
  int b = blockIdx.x;
  float thresh = (*flag) ? 0.85f : 0.5f;
  __shared__ float ssum[256], scnt[256];
  float s = 0.f, c = 0.f;
  for (int n = threadIdx.x; n < NN; n += 256) {
    int id = b * NN + n;
    float fg = (maxiou[id] >= thresh) ? 1.f : 0.f;
    int a = assign[id];
    const float* lg = logits + (size_t)id * KK;
    const float* lb = gtc + ((size_t)b * MM + a) * KK;
    float acc = 0.f;
#pragma unroll 4
    for (int k = 0; k < KK; ++k) {
      float z = lg[k], l = lb[k];
      float t = log1pf(expf(-fabsf(z)));
      float lsp = fminf(z, 0.f) - t;    // log_sigmoid(z)
      float lsn = fminf(-z, 0.f) - t;   // log_sigmoid(-z)
      acc += l * lsp + (1.f - l) * lsn;
    }
    float per = -acc / (float)KK;
    s += per * fg;
    c += fg;
  }
  ssum[threadIdx.x] = s; scnt[threadIdx.x] = c;
  __syncthreads();
  for (int o = 128; o > 0; o >>= 1) {
    if (threadIdx.x < (unsigned)o) {
      ssum[threadIdx.x] += ssum[threadIdx.x + o];
      scnt[threadIdx.x] += scnt[threadIdx.x + o];
    }
    __syncthreads();
  }
  if (threadIdx.x == 0) {
    float cc = scnt[0];
    out[b] = (cc > 0.f) ? ssum[0] / fmaxf(cc, 1.f) : 0.f;
  }
}

// ---------------- launcher ----------------
extern "C" void kernel_launch(void* const* d_in, const int* in_sizes, int n_in,
                              void* d_out, int out_size, void* d_ws, size_t ws_size,
                              hipStream_t stream) {
  const float* feature = (const float*)d_in[0];
  const float* prop    = (const float*)d_in[1];
  const float* gtb     = (const float*)d_in[2];
  const float* gtc     = (const float*)d_in[3];
  const float* W1      = (const float*)d_in[4];
  const float* b1      = (const float*)d_in[5];
  const float* W2      = (const float*)d_in[6];
  const float* b2      = (const float*)d_in[7];
  float* out = (float*)d_out;

  // workspace carve-out (256B aligned blocks)
  char* w = (char*)d_ws;
  auto carve = [&](size_t bytes) {
    char* p = w;
    w += (bytes + 255) & ~(size_t)255;
    return p;
  };
  float*          feat   = (float*)         carve((size_t)BB * HF * WF * CC * 4);   //  1.6 MB
  unsigned short* W1bfT  = (unsigned short*)carve((size_t)KF * HID * 2);            // 51.4 MB (N x K)
  unsigned short* W2bfT  = (unsigned short*)carve((size_t)HID * KK * 2);            //  0.2 MB (N x K)
  unsigned short* pool   = (unsigned short*)carve((size_t)ROWS * KF * 2);           // 102.8 MB
  unsigned short* hidden = (unsigned short*)carve((size_t)ROWS * HID * 2);          //  4.2 MB
  float*          logits = (float*)         carve((size_t)ROWS * KK * 4);           //  0.66 MB
  float*          maxiou = (float*)         carve((size_t)ROWS * 4);
  int*            assign = (int*)           carve((size_t)ROWS * 4);
  int*            flag   = (int*)           carve(256);

  k_tmean<<<(BB * HF * WF * CC + 255) / 256, 256, 0, stream>>>(feature, feat);
  k_cvtT1<<<dim3(HID / 32, KF / 32), dim3(32, 8), 0, stream>>>(W1, W1bfT);
  k_cvtT2<<<(HID * KK + 255) / 256, 256, 0, stream>>>(W2, W2bfT);
  k_iou<<<(ROWS + 255) / 256, 256, 0, stream>>>(prop, gtb, maxiou, assign);
  k_flag<<<1, 256, 0, stream>>>(maxiou, flag);
  k_roipool<<<ROWS, 256, 0, stream>>>(feat, prop, pool);
  k_gemm1<<<dim3(HID / 128, ROWS / 128), 256, 0, stream>>>(pool, W1bfT, b1, hidden);
  k_gemm2<<<(ROWS / 16) * 5 / 8, 256, 0, stream>>>(hidden, W2bfT, b2, logits);
  k_loss<<<BB, 256, 0, stream>>>(logits, gtc, assign, maxiou, flag, out);
}
